// EnhancedTimeSeriesTransformer_29910152249379
// MI455X (gfx1250) — matrix operand order repaired and checked by reading, not compiled, not run on hardware
//
#include <hip/hip_runtime.h>
#include <hip/hip_bf16.h>
#include <math.h>

// ---------------------------------------------------------------------------
// CDNA5 (gfx1250) wave32 WMMA transformer, async global->LDS data movement
// ---------------------------------------------------------------------------
typedef __bf16 bf16;
typedef __attribute__((ext_vector_type(16))) __bf16 v16bf;
typedef __attribute__((ext_vector_type(8)))  __bf16 v8bf;
typedef __attribute__((ext_vector_type(8)))  float  v8f;
typedef __attribute__((ext_vector_type(4)))  int    v4i;

union V16U { v16bf v; v8bf h[2]; };

static constexpr int   CB    = 4;
static constexpr int   CS    = 1024;
static constexpr int   CF    = 64;
static constexpr int   CD    = 512;
static constexpr int   CH    = 8;
static constexpr int   CDK   = 64;
static constexpr int   CDFF  = 2048;
static constexpr int   CL    = 3;
static constexpr int   CREL  = 65;     // 2*MAXREL+1
static constexpr int   MAXREL= 32;
static constexpr float CSCALE= 0.125f; // 1/sqrt(64)
static constexpr float SQRTD = 22.62741699796952f;

// workspace layout (bytes)
static constexpr size_t OFF_H     = 0;
static constexpr size_t OFF_N     = OFF_H     + (size_t)CB*CS*CD*4;      // bf16
static constexpr size_t OFF_Q     = OFF_N     + (size_t)CB*CS*CD*2;
static constexpr size_t OFF_K     = OFF_Q     + (size_t)CB*CS*CD*2;
static constexpr size_t OFF_V     = OFF_K     + (size_t)CB*CS*CD*2;
static constexpr size_t OFF_CTX   = OFF_V     + (size_t)CB*CS*CD*2;
static constexpr size_t OFF_QREL  = OFF_CTX   + (size_t)CB*CS*CD*2;      // f32
static constexpr size_t OFF_FF    = OFF_QREL  + (size_t)CB*CH*CS*CREL*4; // bf16
static constexpr size_t OFF_XS    = OFF_FF    + (size_t)CB*CS*CDFF*2;    // bf16 x*fs
static constexpr size_t OFF_WEMBT = OFF_XS    + (size_t)CB*CS*CF*2;      // bf16 W_emb^T
static constexpr size_t OFF_WT    = OFF_WEMBT + (size_t)CF*CD*2;         // bf16 W^T pool
// per-layer transposed-weight pool layout (elements)
static constexpr size_t WT_Q  = 0;
static constexpr size_t WT_K  = WT_Q + (size_t)CD*CD;
static constexpr size_t WT_V  = WT_K + (size_t)CD*CD;
static constexpr size_t WT_O  = WT_V + (size_t)CD*CD;
static constexpr size_t WT_W1 = WT_O + (size_t)CD*CD;
static constexpr size_t WT_W2 = WT_W1 + (size_t)CD*CDFF;
static constexpr size_t WT_LSTRIDE = WT_W2 + (size_t)CDFF*CD;

// ---------------------------------------------------------------------------
// async global -> LDS copy (16B per lane), ASYNCcnt-tracked
// builtin signature: (v4i AS1* src, v4i AS3* dst, imm offset, imm cpol)
// ---------------------------------------------------------------------------
__device__ __forceinline__ void async_copy16(void* lds, const void* gsrc) {
#if __has_builtin(__builtin_amdgcn_global_load_async_to_lds_b128)
    __builtin_amdgcn_global_load_async_to_lds_b128(
        (__attribute__((address_space(1))) v4i*)gsrc,
        (__attribute__((address_space(3))) v4i*)lds, 0, 0);
#else
    unsigned lds_addr = (unsigned)(unsigned long long)
        (__attribute__((address_space(3))) void*)lds;
    unsigned long long ga = (unsigned long long)gsrc;
    asm volatile("global_load_async_to_lds_b128 %0, %1, off"
                 :: "v"(lds_addr), "v"(ga) : "memory");
#endif
}

template <int N>
__device__ __forceinline__ void wait_asynccnt() {
#if __has_builtin(__builtin_amdgcn_s_wait_asynccnt)
    __builtin_amdgcn_s_wait_asynccnt(N);
#else
    asm volatile("s_wait_asynccnt %0" :: "i"(N) : "memory");
#endif
}

__device__ inline float gelu_exact(float x) {
    return 0.5f * x * (1.0f + erff(x * 0.70710678118654752440f));
}

__device__ inline v16bf ld_frag(const bf16* p0, const bf16* p1) {
    V16U u;
    u.h[0] = *(const v8bf*)p0;
    u.h[1] = *(const v8bf*)p1;
    return u.v;
}

__device__ inline float block_sum_256(float v, float* red) {
    int t = threadIdx.x;
    red[t] = v;
    __syncthreads();
    for (int s = 128; s > 0; s >>= 1) {
        if (t < s) red[t] += red[t + s];
        __syncthreads();
    }
    float r = red[0];
    __syncthreads();
    return r;
}

// ---------------------------------------------------------------------------
// prep: xs = bf16(x * feature_scale)
// ---------------------------------------------------------------------------
__global__ __launch_bounds__(256)
void scale_x_kernel(const float* __restrict__ x, const float* __restrict__ fs,
                    bf16* __restrict__ out)
{
    int idx = blockIdx.x * 256 + threadIdx.x;
    if (idx >= CB * CS * CF) return;
    out[idx] = (bf16)(x[idx] * fs[idx & (CF - 1)]);
}

// prep: W^T[n][k] = bf16(W[k][n])
__global__ __launch_bounds__(256)
void convert_trans_kernel(const float* __restrict__ W, bf16* __restrict__ out,
                          int K, int N)
{
    int idx = blockIdx.x * 256 + threadIdx.x;
    if (idx >= K * N) return;
    int k = idx / N, n = idx - k * N;
    out[(size_t)n * K + k] = (bf16)W[idx];
}

// ---------------------------------------------------------------------------
// LayerNorm (fp32 in -> bf16 out), one 256-thread block per row, D=512
// ---------------------------------------------------------------------------
__global__ __launch_bounds__(256)
void ln_bf16_kernel(const float* __restrict__ h, const float* __restrict__ sc,
                    const float* __restrict__ bi, bf16* __restrict__ out)
{
    __shared__ float red[256];
    int row = blockIdx.x, t = threadIdx.x;
    const float* xr = h + (size_t)row * CD;
    float x0 = xr[t], x1 = xr[t + 256];
    float mean = block_sum_256(x0 + x1, red) * (1.0f / CD);
    float d0 = x0 - mean, d1 = x1 - mean;
    float var = block_sum_256(d0 * d0 + d1 * d1, red) * (1.0f / CD);
    float inv = rsqrtf(var + 1e-6f);
    out[(size_t)row * CD + t]       = (bf16)(d0 * inv * sc[t] + bi[t]);
    out[(size_t)row * CD + t + 256] = (bf16)(d1 * inv * sc[t + 256] + bi[t + 256]);
}

// ---------------------------------------------------------------------------
// generic WMMA GEMM: C[M,N] = act(A_bf16[M,K] @ BT_bf16[N,K]^T + bias)
// 128x64 block tile, BK=32, double-buffered async global->LDS staging.
// ---------------------------------------------------------------------------
#define GBM 128
#define GBN 64
#define GBK 32

__global__ __launch_bounds__(128)
void gemm_bf16_kernel(const bf16* __restrict__ A, const bf16* __restrict__ BT,
                      const float* __restrict__ bias, const float* __restrict__ residual,
                      int res_rows, float f32_scale,
                      float* __restrict__ out_f32, bf16* __restrict__ out_bf16,
                      int M, int N, int K, int gelu_flag, float out_scale)
{
    __shared__ bf16 Alds[2][GBM][GBK];   // 16 KB
    __shared__ bf16 Blds[2][GBN][GBK];   // 8 KB
    const int t    = threadIdx.x;
    const int lane = t & 31;
    const int wave = t >> 5;
    const int wm   = wave >> 1;
    const int wn   = wave & 1;
    const int bm   = blockIdx.y, bn = blockIdx.x;
    const int half = lane >> 4, lm = lane & 15;

    v8f acc[4][2];
    #pragma unroll
    for (int mi = 0; mi < 4; ++mi)
        #pragma unroll
        for (int ni = 0; ni < 2; ++ni)
            acc[mi][ni] = (v8f){0.f,0.f,0.f,0.f,0.f,0.f,0.f,0.f};

    auto issue_tile = [&](int buf, int k0) {
        // A: 128 rows x 32 bf16 = 512 x 16B chunks, 4 per thread
        #pragma unroll
        for (int c = 0; c < 4; ++c) {
            int idx = c * 128 + t;
            int row = idx >> 2, off = (idx & 3) * 8;
            async_copy16(&Alds[buf][row][off],
                         A + (size_t)(bm * GBM + row) * K + k0 + off);
        }
        // B^T: 64 rows x 32 bf16 = 256 x 16B chunks, 2 per thread
        #pragma unroll
        for (int c = 0; c < 2; ++c) {
            int idx = c * 128 + t;
            int row = idx >> 2, off = (idx & 3) * 8;
            async_copy16(&Blds[buf][row][off],
                         BT + (size_t)(bn * GBN + row) * K + k0 + off);
        }
    };

    const int nk = K / GBK;
    issue_tile(0, 0);
    for (int it = 0; it < nk; ++it) {
        const int cur = it & 1;
        if (it + 1 < nk) {
            issue_tile(cur ^ 1, (it + 1) * GBK);
            wait_asynccnt<6>();      // drain current tile, keep next in flight
        } else {
            wait_asynccnt<0>();
        }
        __syncthreads();

        v16bf afrag[4], bfrag[2];
        #pragma unroll
        for (int mi = 0; mi < 4; ++mi) {
            const bf16* p = &Alds[cur][wm * 64 + mi * 16 + lm][0];
            afrag[mi] = ld_frag(p + half * 8, p + 16 + half * 8);
        }
        #pragma unroll
        for (int ni = 0; ni < 2; ++ni) {
            const bf16* p = &Blds[cur][wn * 32 + ni * 16 + lm][0];
            bfrag[ni] = ld_frag(p + half * 8, p + 16 + half * 8);
        }
        #pragma unroll
        for (int mi = 0; mi < 4; ++mi)
            #pragma unroll
            for (int ni = 0; ni < 2; ++ni)
                acc[mi][ni] = __builtin_amdgcn_wmma_f32_16x16x32_bf16(
                    false, afrag[mi], false, bfrag[ni],
                    (short)0, acc[mi][ni], false, false);
        __syncthreads();   // reads of `cur` done before it is refilled at it+2
    }

    // --- epilogue
    #pragma unroll
    for (int mi = 0; mi < 4; ++mi) {
        #pragma unroll
        for (int ni = 0; ni < 2; ++ni) {
            #pragma unroll
            for (int vr = 0; vr < 8; ++vr) {
                int row = bm * GBM + wm * 64 + mi * 16 + vr + 8 * half;
                int col = bn * GBN + wn * 32 + ni * 16 + lm;
                float val = acc[mi][ni][vr] + bias[col];
                if (gelu_flag) val = gelu_exact(val);
                size_t idx = (size_t)row * N + col;
                if (out_f32) {
                    float r = 0.0f;
                    if (residual) {
                        int rrow = res_rows ? (row % res_rows) : row;
                        r = residual[(size_t)rrow * N + col];
                    }
                    out_f32[idx] = r + val * f32_scale;
                }
                if (out_bf16) out_bf16[idx] = (bf16)(val * out_scale);
            }
        }
    }
}

// ---------------------------------------------------------------------------
// qrel[bh,i,r] = q[bh,i,:] . rel_tab[l][r,:]   (q already pre-scaled)
// ---------------------------------------------------------------------------
__global__ __launch_bounds__(256)
void qrel_kernel(const bf16* __restrict__ q, const float* __restrict__ rel_l,
                 float* __restrict__ qrel)
{
    int idx = blockIdx.x * 256 + threadIdx.x;
    if (idx >= CB * CH * CS * CREL) return;
    int r    = idx % CREL;
    int rest = idx / CREL;
    int i    = rest & (CS - 1);
    int bh   = rest >> 10;
    int b = bh >> 3, hh = bh & 7;
    const bf16*  qq = q + ((size_t)(b * CS + i) * CD + hh * CDK);
    const float* rt = rel_l + (size_t)r * CDK;
    float acc = 0.0f;
    #pragma unroll 8
    for (int d = 0; d < CDK; ++d) acc += (float)qq[d] * rt[d];
    qrel[idx] = acc;
}

// ---------------------------------------------------------------------------
// flash attention: block = (b,h, 64 i-rows), 4 waves x 16 rows each.
// K tile staged via async-to-LDS; V transposed manually; P via LDS round-trip
// ---------------------------------------------------------------------------
__global__ __launch_bounds__(128)
void attn_kernel(const bf16* __restrict__ q, const bf16* __restrict__ k,
                 const bf16* __restrict__ v, const float* __restrict__ qrel,
                 bf16* __restrict__ ctx)
{
    __shared__ bf16 Klds[32][64];      // [jj][dk]  4 KB
    __shared__ bf16 Vlds[64][32];      // [dk][jj]  4 KB (transposed)
    __shared__ bf16 Plds[4][16][32];   // per-wave P tile, 4 KB

    const int t = threadIdx.x, lane = t & 31, wave = t >> 5;
    const int half = lane >> 4, lm = lane & 15;
    const int bh = blockIdx.y;
    const int b = bh >> 3, hh = bh & 7;
    const int i0 = blockIdx.x * 64 + wave * 16;
    const int irow = i0 + lm;

    const bf16* qrow = q + ((size_t)(b * CS + irow) * CD + hh * CDK);
    v16bf qfrag[2];
    #pragma unroll
    for (int kc = 0; kc < 2; ++kc)
        qfrag[kc] = ld_frag(qrow + kc * 32 + half * 8, qrow + kc * 32 + 16 + half * 8);

    v8f O[4];
    #pragma unroll
    for (int dkt = 0; dkt < 4; ++dkt)
        O[dkt] = (v8f){0.f,0.f,0.f,0.f,0.f,0.f,0.f,0.f};
    float mrun[8], lrun[8];
    #pragma unroll
    for (int vr = 0; vr < 8; ++vr) { mrun[vr] = -1e30f; lrun[vr] = 0.0f; }

    for (int j0 = 0; j0 < CS; j0 += 32) {
        __syncthreads();
        // K tile: 32 rows x 64 bf16 = 256 x 16B chunks, async, 2 per thread
        #pragma unroll
        for (int c = 0; c < 2; ++c) {
            int idx = c * 128 + t;
            int jj = idx >> 2, off = (idx & 3) * 8;
            async_copy16(&Klds[jj][off],
                         k + ((size_t)(b * CS + j0 + jj) * CD + hh * CDK + off));
        }
        // V tile transposed: manual
        {
            int jj = t >> 2, c0 = (t & 3) * 16;
            const bf16* vs = v + ((size_t)(b * CS + j0 + jj) * CD + hh * CDK + c0);
            #pragma unroll
            for (int e = 0; e < 16; ++e) Vlds[c0 + e][jj] = vs[e];
        }
        wait_asynccnt<0>();
        __syncthreads();

        // scores: two 16x16 j-subtiles, K=64 over two WMMA steps
        v8f s0 = (v8f){0.f,0.f,0.f,0.f,0.f,0.f,0.f,0.f};
        v8f s1 = (v8f){0.f,0.f,0.f,0.f,0.f,0.f,0.f,0.f};
        #pragma unroll
        for (int kc = 0; kc < 2; ++kc) {
            const bf16* p0 = &Klds[lm][kc * 32 + half * 8];
            const bf16* p1 = &Klds[16 + lm][kc * 32 + half * 8];
            v16bf kb0 = ld_frag(p0, p0 + 16);
            v16bf kb1 = ld_frag(p1, p1 + 16);
            s0 = __builtin_amdgcn_wmma_f32_16x16x32_bf16(false, qfrag[kc], false, kb0,
                                                         (short)0, s0, false, false);
            s1 = __builtin_amdgcn_wmma_f32_16x16x32_bf16(false, qfrag[kc], false, kb1,
                                                         (short)0, s1, false, false);
        }

        // bias + online softmax (C-layout: m = vr + 8*half, n = lm)
        float p[2][8], al[8];
        #pragma unroll
        for (int vr = 0; vr < 8; ++vr) {
            int m = vr + 8 * half;
            int ii = i0 + m;
            int j_a = j0 + lm, j_b = j0 + 16 + lm;
            int ra = min(max(j_a - ii, -MAXREL), MAXREL) + MAXREL;
            int rb = min(max(j_b - ii, -MAXREL), MAXREL) + MAXREL;
            const float* qr = qrel + ((size_t)bh * CS + ii) * CREL;
            float v0 = s0[vr] + qr[ra];
            float v1 = s1[vr] + qr[rb];
            float rm = fmaxf(v0, v1);
            #pragma unroll
            for (int off = 1; off < 16; off <<= 1)
                rm = fmaxf(rm, __shfl_xor(rm, off));
            float mn = fmaxf(mrun[vr], rm);
            float a  = __expf(mrun[vr] - mn);
            float e0 = __expf(v0 - mn);
            float e1 = __expf(v1 - mn);
            float rs = e0 + e1;
            #pragma unroll
            for (int off = 1; off < 16; off <<= 1)
                rs += __shfl_xor(rs, off);
            lrun[vr] = lrun[vr] * a + rs;
            mrun[vr] = mn;
            al[vr] = a;
            p[0][vr] = e0;
            p[1][vr] = e1;
        }
        #pragma unroll
        for (int dkt = 0; dkt < 4; ++dkt)
            #pragma unroll
            for (int vr = 0; vr < 8; ++vr)
                O[dkt][vr] *= al[vr];

        // re-layout P (C-layout -> A-fragment) via this wave's LDS slab
        #pragma unroll
        for (int jt = 0; jt < 2; ++jt)
            #pragma unroll
            for (int vr = 0; vr < 8; ++vr)
                Plds[wave][vr + 8 * half][jt * 16 + lm] = (bf16)p[jt][vr];
        asm volatile("s_wait_dscnt 0x0" ::: "memory");

        const bf16* pp = &Plds[wave][lm][0];
        v16bf pf = ld_frag(pp + half * 8, pp + 16 + half * 8);
        #pragma unroll
        for (int dkt = 0; dkt < 4; ++dkt) {
            const bf16* vp = &Vlds[dkt * 16 + lm][half * 8];
            v16bf vf = ld_frag(vp, vp + 16);
            O[dkt] = __builtin_amdgcn_wmma_f32_16x16x32_bf16(false, pf, false, vf,
                                                             (short)0, O[dkt], false, false);
        }
    }

    #pragma unroll
    for (int dkt = 0; dkt < 4; ++dkt) {
        #pragma unroll
        for (int vr = 0; vr < 8; ++vr) {
            int m = vr + 8 * half;
            int ii = i0 + m;
            float val = O[dkt][vr] / lrun[vr];
            ctx[((size_t)(b * CS + ii) * CD) + hh * CDK + dkt * 16 + lm] = (bf16)val;
        }
    }
}

// ---------------------------------------------------------------------------
// final LN (last token only) + classifier head -> logits [B,3]
// ---------------------------------------------------------------------------
__global__ __launch_bounds__(256)
void head_kernel(const float* __restrict__ h, const float* __restrict__ fns,
                 const float* __restrict__ fnb, const float* __restrict__ Wc1,
                 const float* __restrict__ bc1, const float* __restrict__ Wc2,
                 const float* __restrict__ bc2, float* __restrict__ out)
{
    __shared__ float red[256];
    __shared__ float pooled[CD];
    __shared__ float hid[CD / 2];
    int b = blockIdx.x, t = threadIdx.x;
    const float* xr = h + ((size_t)(b * CS + (CS - 1)) * CD);
    float x0 = xr[t], x1 = xr[t + 256];
    float mean = block_sum_256(x0 + x1, red) * (1.0f / CD);
    float d0 = x0 - mean, d1 = x1 - mean;
    float var = block_sum_256(d0 * d0 + d1 * d1, red) * (1.0f / CD);
    float inv = rsqrtf(var + 1e-6f);
    pooled[t]       = d0 * inv * fns[t] + fnb[t];
    pooled[t + 256] = d1 * inv * fns[t + 256] + fnb[t + 256];
    __syncthreads();
    float acc = bc1[t];
    #pragma unroll 8
    for (int d = 0; d < CD; ++d) acc += pooled[d] * Wc1[(size_t)d * (CD / 2) + t];
    hid[t] = gelu_exact(acc);
    __syncthreads();
    if (t < 3) {
        float a = bc2[t];
        for (int j = 0; j < CD / 2; ++j) a += hid[j] * Wc2[(size_t)j * 3 + t];
        out[b * 3 + t] = a;
    }
}

// ---------------------------------------------------------------------------
// launch sequence
// ---------------------------------------------------------------------------
extern "C" void kernel_launch(void* const* d_in, const int* in_sizes, int n_in,
                              void* d_out, int out_size, void* d_ws, size_t ws_size,
                              hipStream_t stream) {
    const float* x       = (const float*)d_in[0];
    const float* fs      = (const float*)d_in[1];
    const float* W_emb   = (const float*)d_in[2];
    const float* b_emb   = (const float*)d_in[3];
    const float* pe      = (const float*)d_in[4];
    const float* Wq      = (const float*)d_in[5];
    const float* bq      = (const float*)d_in[6];
    const float* Wk      = (const float*)d_in[7];
    const float* bk      = (const float*)d_in[8];
    const float* Wv      = (const float*)d_in[9];
    const float* bv      = (const float*)d_in[10];
    const float* Wo      = (const float*)d_in[11];
    const float* bo      = (const float*)d_in[12];
    const float* rel_tab = (const float*)d_in[13];
    const float* ln1_s   = (const float*)d_in[14];
    const float* ln1_b   = (const float*)d_in[15];
    const float* ln2_s   = (const float*)d_in[16];
    const float* ln2_b   = (const float*)d_in[17];
    const float* W1      = (const float*)d_in[18];
    const float* b1      = (const float*)d_in[19];
    const float* W2      = (const float*)d_in[20];
    const float* b2      = (const float*)d_in[21];
    const float* fn_s    = (const float*)d_in[22];
    const float* fn_b    = (const float*)d_in[23];
    const float* Wc1     = (const float*)d_in[24];
    const float* bc1     = (const float*)d_in[25];
    const float* Wc2     = (const float*)d_in[26];
    const float* bc2     = (const float*)d_in[27];

    char* ws = (char*)d_ws;
    float* h     = (float*)(ws + OFF_H);
    bf16*  nbuf  = (bf16*) (ws + OFF_N);
    bf16*  qb    = (bf16*) (ws + OFF_Q);
    bf16*  kb    = (bf16*) (ws + OFF_K);
    bf16*  vb    = (bf16*) (ws + OFF_V);
    bf16*  ctxb  = (bf16*) (ws + OFF_CTX);
    float* qrel  = (float*)(ws + OFF_QREL);
    bf16*  ffb   = (bf16*) (ws + OFF_FF);
    bf16*  xsb   = (bf16*) (ws + OFF_XS);
    bf16*  wembT = (bf16*) (ws + OFF_WEMBT);
    bf16*  wtp   = (bf16*) (ws + OFF_WT);

    const int M = CB * CS;                 // 4096
    const dim3 gemmBlk(128);
    const dim3 gD  (CD   / GBN, M / GBM);  // N=512
    const dim3 gDFF(CDFF / GBN, M / GBM);  // N=2048

    // ---- one-time per-launch weight prep (bf16, transposed) ----
    scale_x_kernel<<<(CB*CS*CF + 255) / 256, 256, 0, stream>>>(x, fs, xsb);
    convert_trans_kernel<<<(CF*CD + 255) / 256, 256, 0, stream>>>(W_emb, wembT, CF, CD);
    for (int l = 0; l < CL; ++l) {
        bf16* wt = wtp + (size_t)l * WT_LSTRIDE;
        const size_t wo = (size_t)l * CD * CD;
        convert_trans_kernel<<<(CD*CD   + 255) / 256, 256, 0, stream>>>(Wq + wo, wt + WT_Q,  CD, CD);
        convert_trans_kernel<<<(CD*CD   + 255) / 256, 256, 0, stream>>>(Wk + wo, wt + WT_K,  CD, CD);
        convert_trans_kernel<<<(CD*CD   + 255) / 256, 256, 0, stream>>>(Wv + wo, wt + WT_V,  CD, CD);
        convert_trans_kernel<<<(CD*CD   + 255) / 256, 256, 0, stream>>>(Wo + wo, wt + WT_O,  CD, CD);
        convert_trans_kernel<<<(CD*CDFF + 255) / 256, 256, 0, stream>>>(W1 + (size_t)l*CD*CDFF, wt + WT_W1, CD, CDFF);
        convert_trans_kernel<<<(CD*CDFF + 255) / 256, 256, 0, stream>>>(W2 + (size_t)l*CDFF*CD, wt + WT_W2, CDFF, CD);
    }

    // ---- embedding as WMMA GEMM: h = sqrtD*(xs @ W_emb + b_emb) + pe ----
    gemm_bf16_kernel<<<gD, gemmBlk, 0, stream>>>(xsb, wembT, b_emb,
        pe, CS, SQRTD, h, nullptr, M, CD, CF, 0, 1.0f);

    for (int l = 0; l < CL; ++l) {
        bf16* wt = wtp + (size_t)l * WT_LSTRIDE;
        // LN1 -> bf16
        ln_bf16_kernel<<<M, 256, 0, stream>>>(h, ln1_s + l*CD, ln1_b + l*CD, nbuf);
        // Q (scale folded), K, V
        gemm_bf16_kernel<<<gD, gemmBlk, 0, stream>>>(nbuf, wt + WT_Q, bq + (size_t)l*CD,
            nullptr, 0, 1.0f, nullptr, qb, M, CD, CD, 0, CSCALE);
        gemm_bf16_kernel<<<gD, gemmBlk, 0, stream>>>(nbuf, wt + WT_K, bk + (size_t)l*CD,
            nullptr, 0, 1.0f, nullptr, kb, M, CD, CD, 0, 1.0f);
        gemm_bf16_kernel<<<gD, gemmBlk, 0, stream>>>(nbuf, wt + WT_V, bv + (size_t)l*CD,
            nullptr, 0, 1.0f, nullptr, vb, M, CD, CD, 0, 1.0f);
        // relative bias table (q pre-scaled, so bias pre-scaled too)
        qrel_kernel<<<(CB*CH*CS*CREL + 255) / 256, 256, 0, stream>>>(
            qb, rel_tab + (size_t)l * CREL * CDK, qrel);
        // flash attention
        attn_kernel<<<dim3(CS / 64, CB * CH), 128, 0, stream>>>(qb, kb, vb, qrel, ctxb);
        // output projection + residual (in-place on h)
        gemm_bf16_kernel<<<gD, gemmBlk, 0, stream>>>(ctxb, wt + WT_O, bo + (size_t)l*CD,
            h, 0, 1.0f, h, nullptr, M, CD, CD, 0, 1.0f);
        // LN2 -> bf16
        ln_bf16_kernel<<<M, 256, 0, stream>>>(h, ln2_s + l*CD, ln2_b + l*CD, nbuf);
        // FFN1 (GELU) -> bf16
        gemm_bf16_kernel<<<gDFF, gemmBlk, 0, stream>>>(nbuf, wt + WT_W1,
            b1 + (size_t)l*CDFF, nullptr, 0, 1.0f, nullptr, ffb, M, CDFF, CD, 1, 1.0f);
        // FFN2 + residual (in-place on h)
        gemm_bf16_kernel<<<gD, gemmBlk, 0, stream>>>(ffb, wt + WT_W2,
            b2 + (size_t)l*CD, h, 0, 1.0f, h, nullptr, M, CD, CDFF, 0, 1.0f);
    }

    head_kernel<<<CB, 256, 0, stream>>>(h, fn_s, fn_b, Wc1, bc1, Wc2, bc2, (float*)d_out);
}
